// ScaledDotProductAttention_48799418417732
// MI455X (gfx1250) — matrix operand order repaired and checked by reading, not compiled, not run on hardware
//
#include <hip/hip_runtime.h>
#include <hip/hip_fp16.h>

// ---- problem constants (match reference setup_inputs) ----
#define BATCH 4
#define NSEQ  2048          // nq == nk == 2048
#define DMODEL 512
#define NHEAD 8
#define DHEAD 64
#define MROWS (BATCH * NSEQ)   // 8192

typedef __attribute__((ext_vector_type(16))) _Float16 v16h;
typedef __attribute__((ext_vector_type(8)))  _Float16 v8h;
typedef __attribute__((ext_vector_type(8)))  float    v8f;

__device__ __forceinline__ v8f wmma_16x16x32(v16h a, v16h b, v8f c) {
  // (neg_a, A, neg_b, B, c_mod, C, reuse_a, reuse_b)
  return __builtin_amdgcn_wmma_f32_16x16x32_f16(false, a, false, b, (short)0, c,
                                                false, false);
}

__device__ __forceinline__ v16h combine8(v8h lo, v8h hi) {
  v16h r;
#pragma unroll
  for (int i = 0; i < 8; ++i) { r[i] = lo[i]; r[8 + i] = hi[i]; }
  return r;
}

// Async copy of 16 bytes/lane from global memory into LDS (ASYNCcnt-tracked).
// VDST VGPR holds the LDS byte offset.
__device__ __forceinline__ void async_b128(unsigned lds_off, const void* gaddr) {
  asm volatile("global_load_async_to_lds_b128 %0, %1, off"
               :: "v"(lds_off), "v"(gaddr)
               : "memory");
}

// ---------------------------------------------------------------------------
// Stage 0: convert f32 weight (K x N, row major) to f16 transposed Wt[n][k]
// ---------------------------------------------------------------------------
__global__ __launch_bounds__(256) void wcvt_kernel(const float* __restrict__ W,
                                                   _Float16* __restrict__ Wt) {
  int idx = blockIdx.x * 256 + threadIdx.x;           // idx = n*512 + k
  if (idx < DMODEL * DMODEL) {
    int n = idx >> 9;
    int k = idx & 511;
    Wt[idx] = (_Float16)W[k * DMODEL + n];
  }
}

// ---------------------------------------------------------------------------
// Stage 1: projection GEMM  X(8192x512) @ W(512x512) + bias
//   mode 0: store rows   -> out[b][h][tok][d]   (Q, K)
//   mode 1: store transp -> out[b][h][d][tok]   (V)
// block: 256 threads = 8 waves, wave computes one 16x16 tile
// grid: (MROWS/16, 4)
// ---------------------------------------------------------------------------
__global__ __launch_bounds__(256) void proj_kernel(
    const float* __restrict__ X, const _Float16* __restrict__ Wt,
    const float* __restrict__ bias, _Float16* __restrict__ out, int mode) {
  const int wave = threadIdx.x >> 5;
  const int lane = threadIdx.x & 31;
  const int row0 = blockIdx.x * 16;
  const int n0 = (blockIdx.y * 8 + wave) * 16;

  const int m = lane & 15;          // A row within tile
  const int half = lane >> 4;       // 0 | 1
  const int aKb = half * 16;        // A halves K offset
  const int koff = half * 16;       // B halves K offset
  const int n = lane & 15;          // B/D column within tile

  v8f acc = {};
  const float* arow = X + (size_t)(row0 + m) * DMODEL;
  const _Float16* brow = Wt + (size_t)(n0 + n) * DMODEL;
#pragma unroll 4
  for (int k0 = 0; k0 < DMODEL; k0 += 32) {
    v16h a;
#pragma unroll
    for (int i = 0; i < 8; ++i) {
      a[i]     = (_Float16)arow[k0 + aKb + i];
      a[8 + i] = (_Float16)arow[k0 + 8 + aKb + i];
    }
    v16h b = *(const v16h*)(brow + k0 + koff);
    acc = wmma_16x16x32(a, b, acc);
  }

  const int col = n0 + n;
  const int h = col >> 6;
  const int d = col & 63;
  const float bv = bias[col];
#pragma unroll
  for (int r = 0; r < 8; ++r) {
    int rowg = row0 + half * 8 + r;
    int b_ = rowg >> 11;            // NSEQ = 2048
    int tok = rowg & (NSEQ - 1);
    float v = acc[r] + bv;
    if (mode == 0)
      out[(((size_t)(b_ * NHEAD + h) * NSEQ) + tok) * DHEAD + d] = (_Float16)v;
    else
      out[(((size_t)(b_ * NHEAD + h) * DHEAD) + d) * NSEQ + tok] = (_Float16)v;
  }
}

// ---------------------------------------------------------------------------
// Stage 2: flash attention.
// block: 256 threads = 8 waves, each wave owns one 16-query tile.
// K/V k-chunks (32 keys) are double-buffered in LDS, staged by wave 0 with
// GLOBAL_LOAD_ASYNC_TO_LDS_B128 (ASYNCcnt), overlapped with WMMA compute.
// grid: (NSEQ/16/8, BATCH*NHEAD)
//
// LDS map (single array, base offset 0):
//   bytes [    0,  4096) kbuf0   [ 4096,  8192) kbuf1     (32 keys x 64d f16)
//   bytes [ 8192, 12288) vbuf0   [12288, 16384) vbuf1     (64d x 32 keys f16)
//   bytes [16384, 24576) per-wave P staging (8 x 1KB)
// ---------------------------------------------------------------------------
__global__ __launch_bounds__(256) void attn_kernel(
    const _Float16* __restrict__ Qh,   // [b][h][nq][64]
    const _Float16* __restrict__ Kh,   // [b][h][nk][64]
    const _Float16* __restrict__ Vt,   // [b][h][64][nk]
    const float* __restrict__ AW,      // [b][h][nq][nk]
    _Float16* __restrict__ O) {        // [b][nq][512]
  __shared__ __align__(64) _Float16 smem[12288];   // 24 KB

  const int wave = threadIdx.x >> 5;
  const int lane = threadIdx.x & 31;
  const int bh = blockIdx.y;
  const int q0 = (blockIdx.x * 8 + wave) * 16;

  const _Float16* Qp = Qh + ((size_t)bh * NSEQ + q0) * DHEAD;
  const char* Kg = (const char*)(Kh + (size_t)bh * NSEQ * DHEAD);
  const char* Vg = (const char*)(Vt + (size_t)bh * DHEAD * NSEQ);
  const float* Wp = AW + ((size_t)bh * NSEQ + q0) * NSEQ;

  const int m = lane & 15;
  const int half = lane >> 4;
  const int aKb = half * 16;
  const int koff = half * 16;
  const int n = lane & 15;

  // Q tile as two A fragments (features 0..31 and 32..63), contiguous loads
  const _Float16* qrow = Qp + (size_t)m * DHEAD;
  v16h qa0 = combine8(*(const v8h*)(qrow + aKb), *(const v8h*)(qrow + 8 + aKb));
  v16h qa1 = combine8(*(const v8h*)(qrow + 32 + aKb), *(const v8h*)(qrow + 40 + aKb));

  v8f o0 = {}, o1 = {}, o2 = {}, o3 = {};
  float mrow[8], lrow[8];
#pragma unroll
  for (int r = 0; r < 8; ++r) { mrow[r] = -3.0e38f; lrow[r] = 0.0f; }

  _Float16* pl = smem + 8192 + wave * 512;

  // prologue: stage chunk 0 into buffer 0
  if (wave == 0) {
    // K chunk: 4 KB contiguous (32 rows x 128B)
#pragma unroll
    for (int j = 0; j < 8; ++j) {
      unsigned off = j * 512 + lane * 16;
      async_b128(off, Kg + off);
    }
    // V chunk: 64 rows x 64 B, global row stride 4096 B
#pragma unroll
    for (int j = 0; j < 8; ++j) {
      int d = j * 8 + (lane >> 2);
      unsigned inrow = (lane & 3) * 16;
      async_b128(16384 /*vbuf0 byte base is 8192*/ - 8192 + (unsigned)(d * 64) + inrow,
                 Vg + (size_t)d * 4096 + inrow);
    }
  }

  for (int k0 = 0; k0 < NSEQ; k0 += 32) {
    const int cur = (k0 >> 5) & 1;
    const int nxt = cur ^ 1;

    if (wave == 0) {
      if (k0 + 32 < NSEQ) {
        // stage next chunk into the other buffer, then wait for current:
        // ASYNCcnt completes in order, so <=16 outstanding == current done.
        const char* Kc = Kg + (size_t)(k0 + 32) * 128;
        const char* Vc = Vg + (size_t)(k0 + 32) * 2;
        const unsigned kb = (unsigned)(nxt * 4096);
        const unsigned vb = (unsigned)(8192 + nxt * 4096);
#pragma unroll
        for (int j = 0; j < 8; ++j) {
          unsigned off = j * 512 + lane * 16;
          async_b128(kb + off, Kc + off);
        }
#pragma unroll
        for (int j = 0; j < 8; ++j) {
          int d = j * 8 + (lane >> 2);
          unsigned inrow = (lane & 3) * 16;
          async_b128(vb + (unsigned)(d * 64) + inrow,
                     Vc + (size_t)d * 4096 + inrow);
        }
        asm volatile("s_wait_asynccnt 16" ::: "memory");
      } else {
        asm volatile("s_wait_asynccnt 0" ::: "memory");
      }
    }
    __syncthreads();   // staged chunk visible to all waves

    const _Float16* kbL = smem + cur * 2048;          // [key][64]
    const _Float16* vbL = smem + 4096 + cur * 2048;   // [d][32]

    // ---- S = Q @ K^T for 16 queries x 32 keys (two 16x16 accumulators) ----
    v8f s0 = {}, s1 = {};
    s0 = wmma_16x16x32(qa0, *(const v16h*)(kbL + (size_t)n * 64 + koff), s0);
    s0 = wmma_16x16x32(qa1, *(const v16h*)(kbL + (size_t)n * 64 + 32 + koff), s0);
    s1 = wmma_16x16x32(qa0, *(const v16h*)(kbL + (size_t)(16 + n) * 64 + koff), s1);
    s1 = wmma_16x16x32(qa1, *(const v16h*)(kbL + (size_t)(16 + n) * 64 + 32 + koff), s1);

    // prefetch next chunk of the 512MB attention-weight stream
    if (k0 + 32 < NSEQ)
      __builtin_prefetch(Wp + (size_t)(half * 8) * NSEQ + k0 + 32, 0, 1);

    // ---- scale by 1/sqrt(64), multiplicative weights, online softmax ----
#pragma unroll
    for (int r = 0; r < 8; ++r) {
      const float* wr = Wp + (size_t)(half * 8 + r) * NSEQ + k0;
      s0[r] = s0[r] * 0.125f * wr[n];
      s1[r] = s1[r] * 0.125f * wr[n + 16];
      float mx = fmaxf(s0[r], s1[r]);
#pragma unroll
      for (int off = 8; off >= 1; off >>= 1)
        mx = fmaxf(mx, __shfl_xor(mx, off, 16));
      float mnew = fmaxf(mrow[r], mx);
      float alpha = __expf(mrow[r] - mnew);
      mrow[r] = mnew;
      float p0 = __expf(s0[r] - mnew);
      float p1 = __expf(s1[r] - mnew);
      float ps = p0 + p1;
#pragma unroll
      for (int off = 8; off >= 1; off >>= 1) ps += __shfl_xor(ps, off, 16);
      lrow[r] = lrow[r] * alpha + ps;
      o0[r] *= alpha; o1[r] *= alpha; o2[r] *= alpha; o3[r] *= alpha;
      s0[r] = p0; s1[r] = p1;
    }

    // ---- stage P (16x32 f16) through per-wave LDS to form an A fragment ----
    // DS ops are in-order within a wave (DScnt), no barrier needed here.
#pragma unroll
    for (int r = 0; r < 8; ++r) {
      int qm = half * 8 + r;
      pl[qm * 32 + n]      = (_Float16)s0[r];
      pl[qm * 32 + 16 + n] = (_Float16)s1[r];
    }
    v16h pa = combine8(*(const v8h*)(pl + m * 32 + aKb),
                       *(const v8h*)(pl + m * 32 + 8 + aKb));

    // ---- O += P @ V from staged LDS V (contiguous B fragments) ----
    o0 = wmma_16x16x32(pa, *(const v16h*)(vbL + (size_t)(n) * 32 + koff), o0);
    o1 = wmma_16x16x32(pa, *(const v16h*)(vbL + (size_t)(16 + n) * 32 + koff), o1);
    o2 = wmma_16x16x32(pa, *(const v16h*)(vbL + (size_t)(32 + n) * 32 + koff), o2);
    o3 = wmma_16x16x32(pa, *(const v16h*)(vbL + (size_t)(48 + n) * 32 + koff), o3);

    __syncthreads();   // all waves done with 'cur' before it is restaged
  }

  // ---- normalize and store f16 to [b][q][h*64 + d] ----
  const int b_ = bh >> 3, h_ = bh & 7;
#pragma unroll
  for (int r = 0; r < 8; ++r) {
    float inv = 1.0f / lrow[r];
    int qm = q0 + half * 8 + r;
    _Float16* orow = O + ((size_t)b_ * NSEQ + qm) * DMODEL + h_ * DHEAD;
    orow[n]      = (_Float16)(o0[r] * inv);
    orow[16 + n] = (_Float16)(o1[r] * inv);
    orow[32 + n] = (_Float16)(o2[r] * inv);
    orow[48 + n] = (_Float16)(o3[r] * inv);
  }
}

// ---------------------------------------------------------------------------
// Stage 3: out = LayerNorm(queries + A @ Wo + bo)
// block: 256 threads (8 waves), 16 rows x 512 cols per block; grid: MROWS/16
// ---------------------------------------------------------------------------
__global__ __launch_bounds__(256) void oproj_ln_kernel(
    const _Float16* __restrict__ A,    // (8192,512) f16 attention output
    const _Float16* __restrict__ WoT,  // Wt[n][k]
    const float* __restrict__ bo, const float* __restrict__ resid,
    const float* __restrict__ ln_g, const float* __restrict__ ln_b,
    float* __restrict__ out) {
  __shared__ float rows[16][DMODEL + 4];

  const int wave = threadIdx.x >> 5;
  const int lane = threadIdx.x & 31;
  const int row0 = blockIdx.x * 16;
  const int m = lane & 15;
  const int half = lane >> 4;
  const int aKb = half * 16;
  const int koff = half * 16;
  const int n = lane & 15;

  const _Float16* arow = A + (size_t)(row0 + m) * DMODEL;
#pragma unroll
  for (int t = 0; t < 4; ++t) {
    const int n0 = (wave * 4 + t) * 16;
    const _Float16* brow = WoT + (size_t)(n0 + n) * DMODEL;
    v8f acc = {};
#pragma unroll 4
    for (int k0 = 0; k0 < DMODEL; k0 += 32) {
      v16h a = combine8(*(const v8h*)(arow + k0 + aKb),
                        *(const v8h*)(arow + k0 + 8 + aKb));
      v16h b = *(const v16h*)(brow + k0 + koff);
      acc = wmma_16x16x32(a, b, acc);
    }
    float bias = bo[n0 + n];
#pragma unroll
    for (int r = 0; r < 8; ++r) {
      int mr = half * 8 + r;
      rows[mr][n0 + n] =
          acc[r] + bias + resid[(size_t)(row0 + mr) * DMODEL + n0 + n];
    }
  }
  __syncthreads();

  // LayerNorm: 16 threads per row (lanes 0-15 / 16-31 are distinct rows)
  const int lr = threadIdx.x >> 4;   // 0..15
  const int lc = threadIdx.x & 15;
  float s1 = 0.0f, s2 = 0.0f;
  for (int c = lc; c < DMODEL; c += 16) {
    float x = rows[lr][c];
    s1 += x; s2 += x * x;
  }
#pragma unroll
  for (int off = 8; off >= 1; off >>= 1) {
    s1 += __shfl_xor(s1, off, 16);
    s2 += __shfl_xor(s2, off, 16);
  }
  const float mu = s1 * (1.0f / DMODEL);
  const float var = s2 * (1.0f / DMODEL) - mu * mu;
  const float rinv = rsqrtf(var + 1e-5f);
  for (int c = lc; c < DMODEL; c += 16) {
    float x = rows[lr][c];
    out[(size_t)(row0 + lr) * DMODEL + c] = (x - mu) * rinv * ln_g[c] + ln_b[c];
  }
}

// ---------------------------------------------------------------------------
extern "C" void kernel_launch(void* const* d_in, const int* in_sizes, int n_in,
                              void* d_out, int out_size, void* d_ws,
                              size_t ws_size, hipStream_t stream) {
  const float* queries = (const float*)d_in[0];
  const float* keys_values = (const float*)d_in[1];
  const float* attw = (const float*)d_in[2];
  const float* Wq = (const float*)d_in[3];
  const float* bq = (const float*)d_in[4];
  const float* Wk = (const float*)d_in[5];
  const float* bk = (const float*)d_in[6];
  const float* Wv = (const float*)d_in[7];
  const float* bv = (const float*)d_in[8];
  const float* Wo = (const float*)d_in[9];
  const float* bo = (const float*)d_in[10];
  const float* ln_g = (const float*)d_in[11];
  const float* ln_b = (const float*)d_in[12];
  float* out = (float*)d_out;

  // workspace carve-up (f16): Qh, Kh, Vt, AttnO each 4.19M elems; 4 weights
  const size_t NTOK = (size_t)BATCH * NSEQ * DMODEL;   // 4,194,304
  const size_t NW = (size_t)DMODEL * DMODEL;           // 262,144
  _Float16* Qh = (_Float16*)d_ws;
  _Float16* Kh = Qh + NTOK;
  _Float16* Vt = Kh + NTOK;
  _Float16* AttnO = Vt + NTOK;
  _Float16* WqT = AttnO + NTOK;
  _Float16* WkT = WqT + NW;
  _Float16* WvT = WkT + NW;
  _Float16* WoT = WvT + NW;

  // Stage 0: weight convert + transpose
  dim3 cb(256), cg((NW + 255) / 256);
  wcvt_kernel<<<cg, cb, 0, stream>>>(Wq, WqT);
  wcvt_kernel<<<cg, cb, 0, stream>>>(Wk, WkT);
  wcvt_kernel<<<cg, cb, 0, stream>>>(Wv, WvT);
  wcvt_kernel<<<cg, cb, 0, stream>>>(Wo, WoT);

  // Stage 1: Q/K/V projections
  dim3 pg(MROWS / 16, 4);
  proj_kernel<<<pg, 256, 0, stream>>>(queries, WqT, bq, Qh, 0);
  proj_kernel<<<pg, 256, 0, stream>>>(keys_values, WkT, bk, Kh, 0);
  proj_kernel<<<pg, 256, 0, stream>>>(keys_values, WvT, bv, Vt, 1);

  // Stage 2: fused weighted-softmax flash attention (async K/V staging)
  dim3 ag(NSEQ / 16 / 8, BATCH * NHEAD);
  attn_kernel<<<ag, 256, 0, stream>>>(Qh, Kh, Vt, attw, AttnO);

  // Stage 3: output projection + residual + LayerNorm
  oproj_ln_kernel<<<MROWS / 16, 256, 0, stream>>>(AttnO, WoT, bo, queries,
                                                  ln_g, ln_b, out);
}
